// Agent_21835613733620
// MI455X (gfx1250) — matrix operand order repaired and checked by reading, not compiled, not run on hardware
//
#include <hip/hip_runtime.h>
#include <hip/hip_bf16.h>

typedef __attribute__((ext_vector_type(16))) __bf16 v16bf;
typedef __attribute__((ext_vector_type(8)))  float  v8f;

#define D_MODEL   256
#define NTILES    16          // 256 / 16
#define LN_EPS    1e-5f
#define CLIPV     5.0f
#define ROWSTRIDE 260         // 256 + 4 pad floats -> conflict-free LDS
#define WPB       4           // waves per block
#define TPB       (WPB * 32)

// ---------------------------------------------------------------- utilities

__global__ void zero_f32_kernel(float* __restrict__ p, int n) {
    int i = blockIdx.x * blockDim.x + threadIdx.x;
    if (i < n) p[i] = 0.0f;
}

__global__ void count_kernel(const int* __restrict__ bidx, int n,
                             float* __restrict__ counts) {
    int i = blockIdx.x * blockDim.x + threadIdx.x;
    if (i < n) atomicAdd(&counts[bidx[i]], 1.0f);
}

// Rearrange a row-major f32 weight [K x 256] into per-(tile,kstep,lane)
// contiguous bf16 B-fragments.  B-matrix 16-bit layout (dense, wave32):
// lane l holds column (l&15); khalf = l>>4; element i holds
// k = s*32 + (i>>3)*16 + khalf*8 + (i&7).
__global__ void prep_weight_kernel(const float* __restrict__ W,
                                   __bf16* __restrict__ out,
                                   int ksteps, int kvalid) {
    int e = blockIdx.x * blockDim.x + threadIdx.x;
    int total = NTILES * ksteps * 32 * 16;
    if (e >= total) return;
    int i = e & 15;
    int l = (e >> 4) & 31;
    int t = e >> 9;              // (n*ksteps + s)
    int s = t % ksteps;
    int n = t / ksteps;
    int col = n * 16 + (l & 15);
    int khalf = l >> 4;
    int k = s * 32 + (i >> 3) * 16 + khalf * 8 + (i & 7);
    float v = (k < kvalid) ? W[(size_t)k * D_MODEL + col] : 0.0f;
    out[e] = (__bf16)v;
}

// ---------------------------------------------------------------- fused core
//
// One wave processes 16 rows:
//   feats -> norm/clamp -> (WMMA embed GEMM) + bias -> relu -> LDS stage
//   -> LayerNorm (in LDS) -> (WMMA backbone GEMM) + bias -> relu
//   -> atomic accumulate into segment sums.
template <int KSTEPS>
__global__ __launch_bounds__(TPB)
void fused_embed_backbone(const float* __restrict__ feats, int nRows, int fK,
                          const float* __restrict__ nmean,
                          const float* __restrict__ nstd,
                          const __bf16* __restrict__ wEmb,   // frag layout
                          const float* __restrict__ bEmb,
                          const float* __restrict__ lng,
                          const float* __restrict__ lnb,
                          const __bf16* __restrict__ wBack,  // frag layout
                          const float* __restrict__ bBack,
                          const int* __restrict__ bidx, int bidxOff,
                          float* __restrict__ sums) {
    __shared__ float smem[WPB * 16 * ROWSTRIDE];

    const int lane  = threadIdx.x & 31;
    const int wave  = threadIdx.x >> 5;
    const int m16   = lane & 15;   // row-in-tile for A-frags / col for B-frags
    const int khalf = lane >> 4;

    const int nTiles = nRows >> 4;             // nRows is a multiple of 16
    int  tile   = blockIdx.x * WPB + wave;
    bool active = (tile < nTiles);
    if (!active) tile = 0;                     // clamp: uniform control flow
    const int rowBase = tile * 16;

    float* stage = smem + wave * 16 * ROWSTRIDE;

    // ---- stage 1: build normalized bf16 A-fragments from feats -----------
    // A-matrix 16-bit layout: lane's 16 elements are two contiguous k-runs:
    //   run0: k = s*32 + khalf*8 + [0..8)
    //   run1: k = s*32 + 16 + khalf*8 + [0..8)
    v16bf aF[KSTEPS];
    {
        const float* frow = feats + (size_t)(rowBase + m16) * fK;
#pragma unroll
        for (int s = 0; s < KSTEPS; ++s) {
#pragma unroll
            for (int run = 0; run < 2; ++run) {
                int k0 = s * 32 + run * 16 + khalf * 8;
#pragma unroll
                for (int j = 0; j < 8; ++j) {
                    int k = k0 + j;
                    float v = 0.0f;
                    if (k < fK) {
                        v = (frow[k] - nmean[k]) / nstd[k];
                        v = fminf(fmaxf(v, -CLIPV), CLIPV);
                    }
                    aF[s][run * 8 + j] = (__bf16)v;
                }
            }
        }
    }

    // ---- embed GEMM: 16 output tiles, bias + relu, stage to LDS ----------
#pragma unroll
    for (int n = 0; n < NTILES; ++n) {
        v8f acc = {};
#pragma unroll
        for (int s = 0; s < KSTEPS; ++s) {
            const v16bf bF =
                *(const v16bf*)(wEmb + ((size_t)(n * KSTEPS + s) * 32 + lane) * 16);
            acc = __builtin_amdgcn_wmma_f32_16x16x32_bf16(
                false, aF[s], false, bF, (short)0, acc, false, false);
        }
        float bias = bEmb[n * 16 + m16];
#pragma unroll
        for (int r = 0; r < 8; ++r) {
            int m = r + khalf * 8;             // C-tile: VGPR r -> M = r + 8*khalf
            stage[m * ROWSTRIDE + n * 16 + m16] = fmaxf(acc[r] + bias, 0.0f);
        }
    }
    __syncthreads();

    // ---- LayerNorm: lane pair (l, l+16) handles row m16 ------------------
    {
        float* rp = stage + m16 * ROWSTRIDE + khalf * 128;
        float sum = 0.0f, sq = 0.0f;
#pragma unroll
        for (int c = 0; c < 128; c += 4) {
            float4 v = *(const float4*)(rp + c);
            sum += v.x + v.y + v.z + v.w;
            sq  += v.x * v.x + v.y * v.y + v.z * v.z + v.w * v.w;
        }
        sum += __shfl_xor(sum, 16, 32);
        sq  += __shfl_xor(sq, 16, 32);
        float mu   = sum * (1.0f / 256.0f);
        float var  = sq * (1.0f / 256.0f) - mu * mu;
        float rstd = rsqrtf(var + LN_EPS);
#pragma unroll
        for (int c = 0; c < 128; ++c) {
            int col = khalf * 128 + c;
            rp[c] = (rp[c] - mu) * rstd * lng[col] + lnb[col];
        }
    }
    __syncthreads();

    // ---- backbone A-fragments from LDS (8 k-steps, hoisted) --------------
    v16bf xF[8];
    {
        const float* rp = stage + m16 * ROWSTRIDE;
#pragma unroll
        for (int s = 0; s < 8; ++s) {
#pragma unroll
            for (int run = 0; run < 2; ++run) {
                int k0 = s * 32 + run * 16 + khalf * 8;
                float4 v0 = *(const float4*)(rp + k0);
                float4 v1 = *(const float4*)(rp + k0 + 4);
                xF[s][run * 8 + 0] = (__bf16)v0.x;
                xF[s][run * 8 + 1] = (__bf16)v0.y;
                xF[s][run * 8 + 2] = (__bf16)v0.z;
                xF[s][run * 8 + 3] = (__bf16)v0.w;
                xF[s][run * 8 + 4] = (__bf16)v1.x;
                xF[s][run * 8 + 5] = (__bf16)v1.y;
                xF[s][run * 8 + 6] = (__bf16)v1.z;
                xF[s][run * 8 + 7] = (__bf16)v1.w;
            }
        }
    }

    // segment index per C-tile row (fixed across tiles); -1 => no store
    int bI[8];
#pragma unroll
    for (int r = 0; r < 8; ++r) {
        int row = rowBase + r + khalf * 8;
        bI[r] = (active && row < nRows) ? bidx[bidxOff + row] : -1;
    }

    // ---- backbone GEMM + bias + relu + atomic segment-sum ----------------
#pragma unroll
    for (int n = 0; n < NTILES; ++n) {
        v8f acc = {};
#pragma unroll
        for (int s = 0; s < 8; ++s) {
            const v16bf bF =
                *(const v16bf*)(wBack + ((size_t)(n * 8 + s) * 32 + lane) * 16);
            acc = __builtin_amdgcn_wmma_f32_16x16x32_bf16(
                false, xF[s], false, bF, (short)0, acc, false, false);
        }
        float bias = bBack[n * 16 + m16];
        int col = n * 16 + m16;
#pragma unroll
        for (int r = 0; r < 8; ++r) {
            float v = fmaxf(acc[r] + bias, 0.0f);
            if (bI[r] >= 0)
                atomicAdd(&sums[(size_t)bI[r] * D_MODEL + col], v);
        }
    }
}

// ---------------------------------------------------------------- final proj
__global__ void final_kernel(const float* __restrict__ sums,
                             const float* __restrict__ counts,
                             const float* __restrict__ Wv,
                             const float* __restrict__ bv,
                             float* __restrict__ out, int B) {
    int b = blockIdx.x * blockDim.x + threadIdx.x;
    if (b >= B) return;
    float inv = 1.0f / fmaxf(counts[b], 1.0f);
    const float* row = sums + (size_t)b * D_MODEL;
    float acc = 0.0f;
#pragma unroll 4
    for (int k = 0; k < D_MODEL; ++k) acc += row[k] * inv * Wv[k];
    out[b] = acc + bv[0];
}

// ---------------------------------------------------------------- launcher
extern "C" void kernel_launch(void* const* d_in, const int* in_sizes, int n_in,
                              void* d_out, int out_size, void* d_ws, size_t ws_size,
                              hipStream_t stream) {
    const float* featsA = (const float*)d_in[0];
    const float* featsB = (const float*)d_in[1];
    const float* nAm    = (const float*)d_in[2];
    const float* nAs    = (const float*)d_in[3];
    const float* nBm    = (const float*)d_in[4];
    const float* nBs    = (const float*)d_in[5];
    const float* WA     = (const float*)d_in[6];
    const float* bA     = (const float*)d_in[7];
    const float* lnAg   = (const float*)d_in[8];
    const float* lnAb   = (const float*)d_in[9];
    const float* WB_    = (const float*)d_in[10];
    const float* bB     = (const float*)d_in[11];
    const float* lnBg   = (const float*)d_in[12];
    const float* lnBb   = (const float*)d_in[13];
    const float* Wback  = (const float*)d_in[14];
    const float* bback  = (const float*)d_in[15];
    const float* Wv     = (const float*)d_in[16];
    const float* bv     = (const float*)d_in[17];
    const int*   bidx   = (const int*)d_in[18];
    float*       out    = (float*)d_out;

    const int fA = in_sizes[2];
    const int fB = in_sizes[4];
    const int nA = in_sizes[0] / fA;
    const int nB = in_sizes[1] / fB;
    const int B  = out_size;          // num_obs

    // workspace layout
    char*  ws     = (char*)d_ws;
    size_t off    = 0;
    float* sums   = (float*)(ws + off); off += (size_t)B * D_MODEL * 4;
    off = (off + 255) & ~(size_t)255;
    float* counts = (float*)(ws + off); off += (size_t)B * 4;
    off = (off + 255) & ~(size_t)255;
    __bf16* wsWA    = (__bf16*)(ws + off); off += (size_t)NTILES * 1 * 32 * 16 * 2;
    __bf16* wsWB    = (__bf16*)(ws + off); off += (size_t)NTILES * 2 * 32 * 16 * 2;
    __bf16* wsWback = (__bf16*)(ws + off); off += (size_t)NTILES * 8 * 32 * 16 * 2;
    (void)ws_size; (void)n_in;

    // zero accumulators
    int znS = B * D_MODEL;
    zero_f32_kernel<<<(znS + 255) / 256, 256, 0, stream>>>(sums, znS);
    zero_f32_kernel<<<(B + 255) / 256, 256, 0, stream>>>(counts, B);

    // weight fragment prep (bf16)
    int eA  = NTILES * 1 * 32 * 16;
    int eB  = NTILES * 2 * 32 * 16;
    int eBk = NTILES * 8 * 32 * 16;
    prep_weight_kernel<<<(eA  + 255) / 256, 256, 0, stream>>>(WA,    wsWA,    1, fA);
    prep_weight_kernel<<<(eB  + 255) / 256, 256, 0, stream>>>(WB_,   wsWB,    2, fB);
    prep_weight_kernel<<<(eBk + 255) / 256, 256, 0, stream>>>(Wback, wsWback, 8, D_MODEL);

    // segment counts
    count_kernel<<<(nA + nB + 255) / 256, 256, 0, stream>>>(bidx, nA + nB, counts);

    // fused embed + backbone + pooled accumulation
    int tilesA  = (nA + 15) / 16;
    int tilesB  = (nB + 15) / 16;
    int blocksA = (tilesA + WPB - 1) / WPB;
    int blocksB = (tilesB + WPB - 1) / WPB;
    fused_embed_backbone<1><<<blocksA, TPB, 0, stream>>>(
        featsA, nA, fA, nAm, nAs, wsWA, bA, lnAg, lnAb,
        wsWback, bback, bidx, 0, sums);
    fused_embed_backbone<2><<<blocksB, TPB, 0, stream>>>(
        featsB, nB, fB, nBm, nBs, wsWB, bB, lnBg, lnBb,
        wsWback, bback, bidx, nA, sums);

    // final projection
    final_kernel<<<(B + 255) / 256, 256, 0, stream>>>(sums, counts, Wv, bv, out, B);
}
